// new_simHGCL_52458730553657
// MI455X (gfx1250) — matrix operand order repaired and checked by reading, not compiled, not run on hardware
//
#include <hip/hip_runtime.h>
#include <hip/hip_fp16.h>

typedef __attribute__((ext_vector_type(16))) _Float16 v16h;
typedef __attribute__((ext_vector_type(8)))  float    v8f;

#define FD 64
#define HID 128

// ---------------- degree / scale kernels ----------------

__global__ void count_kernel(const int* __restrict__ idx, int E, int* __restrict__ cnt) {
  int i = blockIdx.x * blockDim.x + threadIdx.x;
  if (i < E) atomicAdd(&cnt[idx[i]], 1);
}

// gcn: dinv = deg>0 ? rsqrt(deg) : 0
__global__ void gcn_scale_kernel(const int* __restrict__ cnt, float* __restrict__ s, int n) {
  int i = blockIdx.x * blockDim.x + threadIdx.x;
  if (i < n) {
    int c = cnt[i];
    s[i] = (c > 0) ? rsqrtf((float)c) : 0.0f;
  }
}

// graph_conv: rsqrt(max(deg,1))
__global__ void conv_scale_kernel(const int* __restrict__ cnt, float* __restrict__ s, int n) {
  int i = blockIdx.x * blockDim.x + threadIdx.x;
  if (i < n) s[i] = rsqrtf(fmaxf((float)cnt[i], 1.0f));
}

// ---------------- SpMM: one wave32 per edge, float2 per lane ----------------

__global__ void spmm_kernel(const int* __restrict__ src, const int* __restrict__ dst, int E,
                            const float* __restrict__ x, const float* __restrict__ ssrc,
                            float* __restrict__ out) {
  int wave = blockIdx.x * (blockDim.x >> 5) + (threadIdx.x >> 5);
  int lane = threadIdx.x & 31;
  if (wave >= E) return;
  int s = src[wave];
  int d = dst[wave];
  float sc = ssrc[s];
  const float2* xp = (const float2*)(x + (size_t)s * FD);
  float2 v = xp[lane];
  float* op = out + (size_t)d * FD + lane * 2;
  atomicAdd(op,     v.x * sc);
  atomicAdd(op + 1, v.y * sc);
}

__global__ void postscale_kernel(float* __restrict__ x, const float* __restrict__ dsc, int n) {
  int i = blockIdx.x * blockDim.x + threadIdx.x;
  if (i < n * FD) x[i] *= dsc[i / FD];
}

// ---------------- branchless tanh (no EXEC churn inside WMMA loop) ----------------
__device__ __forceinline__ float fast_tanh(float x) {
  float xc = fminf(fmaxf(x, -10.0f), 10.0f);   // tanh saturated well before +-10
  float e  = __expf(2.0f * xc);                // single v_exp_f32
  return (e - 1.0f) / (e + 1.0f);
}

// ---------------- semantic attention score via WMMA ----------------
// One wave per 16-node tile: P = Z(16x64) @ W1(64x128), score = sum_n sum_col tanh(P+b1)*w2
// W1 staged once per workgroup in LDS, transposed + f16: w1t[n][k] (16 KB of the 320 KB WGP LDS)
__global__ void attn_kernel(const float* __restrict__ z, const float* __restrict__ W1,
                            const float* __restrict__ bias, const float* __restrict__ wv2,
                            float* __restrict__ wsum, int n) {
  __shared__ __align__(32) _Float16 w1t[HID * FD];   // [col][k]
  for (int idx = threadIdx.x; idx < HID * FD; idx += blockDim.x) {
    int col = idx >> 6;          // 0..127
    int k   = idx & 63;          // 0..63
    w1t[idx] = (_Float16)W1[k * HID + col];
  }
  __syncthreads();

  int wave = blockIdx.x * (blockDim.x >> 5) + (threadIdx.x >> 5);
  int lane = threadIdx.x & 31;
  int node_base = wave * 16;
  if (node_base >= n) return;           // wave-uniform: EXEC stays all-ones for WMMA
  int half_sel = lane >> 4;
  int r16      = lane & 15;

  // A fragments (contiguous k order used here, not the ISA table order): lanes 0-15 rows hold K0..15;
  // lanes 16-31: K16..31 (matches B). Two K-blocks cover K=64.
  const float* zrow = z + (size_t)(node_base + r16) * FD;
  v16h a0, a1;
#pragma unroll
  for (int j = 0; j < 8; ++j) {
    a0[j]     = (_Float16)zrow[half_sel * 16 + j];
    a0[j + 8] = (_Float16)zrow[half_sel * 16 + 8 + j];
    a1[j]     = (_Float16)zrow[32 + half_sel * 16 + j];
    a1[j + 8] = (_Float16)zrow[32 + half_sel * 16 + 8 + j];
  }

  float tot = 0.0f;
#pragma unroll
  for (int ct = 0; ct < HID / 16; ++ct) {
    int col = ct * 16 + r16;            // column N of W1 for this lane
    // B fragments (16-bit B 32x16): lanes 0-15 hold K0..15, lanes 16-31 hold K16..31.
    // Transposed LDS layout makes each fragment one contiguous 32B chunk.
    const v16h* bp = (const v16h*)(w1t + col * FD);
    v16h bf0 = bp[half_sel];            // K = half_sel*16 + (0..15)
    v16h bf1 = bp[2 + half_sel];        // K = 32 + half_sel*16 + (0..15)
    v8f c = {};
    c = __builtin_amdgcn_wmma_f32_16x16x32_f16(false, a0, false, bf0, (short)0, c, false, false);
    c = __builtin_amdgcn_wmma_f32_16x16x32_f16(false, a1, false, bf1, (short)0, c, false, false);
    float bb = bias[col];
    float ww = wv2[col];
#pragma unroll
    for (int r = 0; r < 8; ++r)         // c[r] = P[row = r + 8*half_sel][col]
      tot += fast_tanh(c[r] + bb) * ww;
  }
  // reduce over the wave (16 rows x all cols), one atomic per wave
#pragma unroll
  for (int off = 16; off > 0; off >>= 1) tot += __shfl_xor(tot, off, 32);
  if (lane == 0) atomicAdd(wsum, tot);
}

// ---------------- beta softmax + weighted combine ----------------

__global__ void combine_kernel(const float* __restrict__ z1, const float* __restrict__ z2,
                               const float* __restrict__ wsum, float* __restrict__ out, int n) {
  int i = blockIdx.x * blockDim.x + threadIdx.x;
  if (i >= n * FD) return;
  float m0 = wsum[0] / (float)n;
  float m1 = wsum[1] / (float)n;
  float mx = fmaxf(m0, m1);
  float e0 = expf(m0 - mx), e1 = expf(m1 - mx);
  float inv = 1.0f / (e0 + e1);
  out[i] = (e0 * inv) * z1[i] + (e1 * inv) * z2[i];
}

// ---------------- final blend + gathers ----------------

__global__ void gather_kernel(const float* __restrict__ hu, const float* __restrict__ hi,
                              const float* __restrict__ ui,
                              const int* __restrict__ uidx, const int* __restrict__ iidx,
                              const int* __restrict__ nidx,
                              float* __restrict__ out, int Un, int Bn) {
  int i = blockIdx.x * blockDim.x + threadIdx.x;
  int total = 3 * Bn * FD;
  if (i >= total) return;
  int sec = i / (Bn * FD);
  int r   = (i / FD) % Bn;
  int d   = i % FD;
  float v;
  if (sec == 0) {
    int u = uidx[r];
    v = 0.5f * hu[(size_t)u * FD + d] + 0.5f * ui[(size_t)u * FD + d];
  } else {
    int rr = (sec == 1) ? r : nidx[r];
    int it = iidx[rr];
    v = 0.5f * hi[(size_t)it * FD + d] + 0.5f * ui[(size_t)(Un + it) * FD + d];
  }
  out[i] = v;
}

// ---------------- host orchestration ----------------

static inline int cdiv(long long a, long long b) { return (int)((a + b - 1) / b); }

extern "C" void kernel_launch(void* const* d_in, const int* in_sizes, int n_in,
                              void* d_out, int out_size, void* d_ws, size_t ws_size,
                              hipStream_t stream) {
  const float* user_feat = (const float*)d_in[0];
  const float* item_feat = (const float*)d_in[1];
  const float* sa_u_W1   = (const float*)d_in[2];
  const float* sa_u_b1   = (const float*)d_in[3];
  const float* sa_u_w2   = (const float*)d_in[4];
  const float* sa_i_W1   = (const float*)d_in[5];
  const float* sa_i_b1   = (const float*)d_in[6];
  const float* sa_i_w2   = (const float*)d_in[7];
  const int*   ui_edges  = (const int*)d_in[8];
  const int*   u_e1      = (const int*)d_in[9];
  const int*   u_e2      = (const int*)d_in[10];
  const int*   i_e1      = (const int*)d_in[11];
  const int*   i_e2      = (const int*)d_in[12];
  const int*   user_idx  = (const int*)d_in[13];
  const int*   item_idx  = (const int*)d_in[14];
  const int*   neg_idx   = (const int*)d_in[15];

  const int Un  = in_sizes[0] / FD;     // 50000
  const int In  = in_sizes[1] / FD;     // 100000
  const int Nn  = Un + In;              // 150000
  const int EUI = in_sizes[8] / 2;      // 2,000,000 (symmetrized)
  const int EU  = in_sizes[9] / 2;      // 500,000
  const int EI  = in_sizes[11] / 2;     // 1,000,000
  const int Bn  = in_sizes[13];         // 8192

  // ---- workspace carve-out ----
  char* ws = (char*)d_ws;
  size_t off = 0;
  auto carve = [&](size_t bytes) -> char* {
    char* p = ws + off;
    off += (bytes + 255) & ~(size_t)255;
    return p;
  };
  float* ui_a = (float*)carve((size_t)Nn * FD * 4);
  float* ui_b = (float*)carve((size_t)Nn * FD * 4);
  float* hu   = (float*)carve((size_t)Un * FD * 4);
  float* hi   = (float*)carve((size_t)In * FD * 4);
  float* zu1  = (float*)carve((size_t)Un * FD * 4);
  float* zu2  = (float*)carve((size_t)Un * FD * 4);
  float* zi1  = (float*)carve((size_t)In * FD * 4);
  float* zi2  = (float*)carve((size_t)In * FD * 4);
  float* gcn_dinv = (float*)carve((size_t)Nn * 4);
  float* u1s = (float*)carve((size_t)Un * 4);
  float* u1d = (float*)carve((size_t)Un * 4);
  float* u2s = (float*)carve((size_t)Un * 4);
  float* u2d = (float*)carve((size_t)Un * 4);
  float* i1s = (float*)carve((size_t)In * 4);
  float* i1d = (float*)carve((size_t)In * 4);
  float* i2s = (float*)carve((size_t)In * 4);
  float* i2d = (float*)carve((size_t)In * 4);
  int*   cntA = (int*)carve((size_t)Nn * 4);
  int*   cntB = (int*)carve((size_t)Nn * 4);
  float* wsum = (float*)carve(4 * sizeof(float));

  const int TB = 256;

  // ---- degrees (edges constant across both iterations) ----
  hipMemsetAsync(cntA, 0, (size_t)Nn * 4, stream);
  count_kernel<<<cdiv(EUI, TB), TB, 0, stream>>>(ui_edges /*row*/, EUI, cntA);
  gcn_scale_kernel<<<cdiv(Nn, TB), TB, 0, stream>>>(cntA, gcn_dinv, Nn);

  auto conv_deg = [&](const int* e, int E, int n, float* sinv, float* dinv) {
    hipMemsetAsync(cntA, 0, (size_t)n * 4, stream);
    hipMemsetAsync(cntB, 0, (size_t)n * 4, stream);
    count_kernel<<<cdiv(E, TB), TB, 0, stream>>>(e,     E, cntA);  // src degrees
    count_kernel<<<cdiv(E, TB), TB, 0, stream>>>(e + E, E, cntB);  // dst degrees
    conv_scale_kernel<<<cdiv(n, TB), TB, 0, stream>>>(cntA, sinv, n);
    conv_scale_kernel<<<cdiv(n, TB), TB, 0, stream>>>(cntB, dinv, n);
  };
  conv_deg(u_e1, EU, Un, u1s, u1d);
  conv_deg(u_e2, EU, Un, u2s, u2d);
  conv_deg(i_e1, EI, In, i1s, i1d);
  conv_deg(i_e2, EI, In, i2s, i2d);

  // ---- init states ----
  hipMemcpyAsync(ui_a,                    user_feat, (size_t)Un * FD * 4, hipMemcpyDeviceToDevice, stream);
  hipMemcpyAsync(ui_a + (size_t)Un * FD,  item_feat, (size_t)In * FD * 4, hipMemcpyDeviceToDevice, stream);
  hipMemcpyAsync(hu,                      user_feat, (size_t)Un * FD * 4, hipMemcpyDeviceToDevice, stream);
  hipMemcpyAsync(hi,                      item_feat, (size_t)In * FD * 4, hipMemcpyDeviceToDevice, stream);

  auto conv = [&](const int* e, int E, int n, const float* x,
                  const float* sinv, const float* dinv, float* z) {
    hipMemsetAsync(z, 0, (size_t)n * FD * 4, stream);
    spmm_kernel<<<cdiv(E, 8), TB, 0, stream>>>(e /*src*/, e + E /*dst*/, E, x, sinv, z);
    postscale_kernel<<<cdiv((long long)n * FD, TB), TB, 0, stream>>>(z, dinv, n);
  };

  float* ui_src = ui_a;
  float* ui_dst = ui_b;
  for (int it = 0; it < 2; ++it) {
    // --- gcn_spmm over bipartite graph: gather from col, scatter to row ---
    hipMemsetAsync(ui_dst, 0, (size_t)Nn * FD * 4, stream);
    spmm_kernel<<<cdiv(EUI, 8), TB, 0, stream>>>(ui_edges + EUI /*col*/, ui_edges /*row*/,
                                                 EUI, ui_src, gcn_dinv, ui_dst);
    postscale_kernel<<<cdiv((long long)Nn * FD, TB), TB, 0, stream>>>(ui_dst, gcn_dinv, Nn);

    // --- HAN users ---
    conv(u_e1, EU, Un, hu, u1s, u1d, zu1);
    conv(u_e2, EU, Un, hu, u2s, u2d, zu2);
    hipMemsetAsync(wsum, 0, 4 * sizeof(float), stream);
    attn_kernel<<<cdiv(Un / 16, 8), TB, 0, stream>>>(zu1, sa_u_W1, sa_u_b1, sa_u_w2, wsum + 0, Un);
    attn_kernel<<<cdiv(Un / 16, 8), TB, 0, stream>>>(zu2, sa_u_W1, sa_u_b1, sa_u_w2, wsum + 1, Un);
    combine_kernel<<<cdiv((long long)Un * FD, TB), TB, 0, stream>>>(zu1, zu2, wsum, hu, Un);

    // --- HAN items ---
    conv(i_e1, EI, In, hi, i1s, i1d, zi1);
    conv(i_e2, EI, In, hi, i2s, i2d, zi2);
    attn_kernel<<<cdiv(In / 16, 8), TB, 0, stream>>>(zi1, sa_i_W1, sa_i_b1, sa_i_w2, wsum + 2, In);
    attn_kernel<<<cdiv(In / 16, 8), TB, 0, stream>>>(zi2, sa_i_W1, sa_i_b1, sa_i_w2, wsum + 3, In);
    combine_kernel<<<cdiv((long long)In * FD, TB), TB, 0, stream>>>(zi1, zi2, wsum + 2, hi, In);

    float* t = ui_src; ui_src = ui_dst; ui_dst = t;
  }

  // ---- final blend + gathers (ui result now in ui_src) ----
  gather_kernel<<<cdiv((long long)3 * Bn * FD, TB), TB, 0, stream>>>(
      hu, hi, ui_src, user_idx, item_idx, neg_idx, (float*)d_out, Un, Bn);
}
